// SortedModel_89326729822602
// MI455X (gfx1250) — matrix operand order repaired and checked by reading, not compiled, not run on hardware
//
#include <hip/hip_runtime.h>

#define BATCH 32768
#define MAXL  9
#define EMBD  512
#define HIDD  256
#define VOCAB 4094

typedef __attribute__((ext_vector_type(16))) __bf16 vbf16;
typedef __attribute__((ext_vector_type(8)))  __bf16 vbf8;
typedef __attribute__((ext_vector_type(8)))  float  vf8;

// D = A*B + C, 16x16x32 bf16 -> f32, wave32
#define WMMA_BF16(A,B,C) \
  __builtin_amdgcn_wmma_f32_16x16x32_bf16(false,(A),false,(B),(short)0,(C),false,false)

__device__ __forceinline__ vf8 vzero8() {
  vf8 z = {0.f,0.f,0.f,0.f,0.f,0.f,0.f,0.f};
  return z;
}

// A fragment (16x32 bf16): lane half h, row M = lane&15.
// Per lane: K = kb..kb+7 (VGPR0-3) and K = kb+16..kb+23 (VGPR4-7), kb = kt*32 + h*8.
__device__ __forceinline__ vbf16 load_a_bf16(const __bf16* p) {
  union { vbf16 v; vbf8 h[2]; } u;
  u.h[0] = *(const vbf8*)(p);
  u.h[1] = *(const vbf8*)(p + 16);
  return u.v;
}

// B fragment (32x16 bf16): col N = lane&15; per lane K = kb..kb+7 and kb+16..kb+23, kb = kt*32 + h*8.
__device__ __forceinline__ vbf16 load_b_bf16(const __bf16* p) {
  union { vbf16 v; vbf8 h[2]; } u;
  u.h[0] = *(const vbf8*)(p);
  u.h[1] = *(const vbf8*)(p + 16);
  return u.v;
}

// A fragment built from f32 source (convert on the fly)
__device__ __forceinline__ vbf16 load_a_f32(const float* p) {
  vbf16 v;
  #pragma unroll
  for (int i = 0; i < 8; ++i) { v[i] = (__bf16)p[i]; v[8 + i] = (__bf16)p[16 + i]; }
  return v;
}

__device__ __forceinline__ float sigmoidf_(float x) { return 1.f / (1.f + __expf(-x)); }

// ---------------------------------------------------------------- prep: f32 -> bf16
__global__ void cvt_bf16(const float* __restrict__ s, __bf16* __restrict__ d, int n) {
  int i = blockIdx.x * blockDim.x + threadIdx.x;
  int stride = gridDim.x * blockDim.x;
  for (; i < n; i += stride) d[i] = (__bf16)s[i];
}

// ---------------------------------------------------------------- fused attention + combine + GRU
struct Smem {
  __bf16 xA[16 * (EMBD + HIDD)];   // [16 rows][768] combine input, bf16
  float  hidH[16 * HIDD];          // hidden tile, f32
  union {
    __bf16 x2[16 * EMBD];          // relu(combine) [16][512] bf16 (phases 2-3)
    float  encS[MAXL * HIDD];      // one row's encoder output (phase 1)
  } u;
  float s1[81];
  float s2[MAXL];
  float w[81];
  float alpha[MAXL];
};

__global__ __launch_bounds__(32) void fused_attn_gru(
    const int* __restrict__ dw,  const float* __restrict__ hid,
    const float* __restrict__ enc, const float* __restrict__ emb,
    const float* __restrict__ aW,  const float* __restrict__ aB,
    const __bf16* __restrict__ wc,  const float* __restrict__ cb,
    const __bf16* __restrict__ wih, const float* __restrict__ bih,
    const __bf16* __restrict__ whh, const float* __restrict__ bhh,
    float* __restrict__ hout, __bf16* __restrict__ hbf)
{
  __shared__ Smem sm;
  const int lane = threadIdx.x;
  const int half = lane >> 4;
  const int l15  = lane & 15;
  const long r0  = (long)blockIdx.x * 16;

  // ---- Phase 1: attention + embedding -> xA[16][768] bf16, hidH[16][256] f32
  for (int r = 0; r < 16; ++r) {
    const long row = r0 + r;
    {
      const float4* e4 = (const float4*)(enc + row * (MAXL * HIDD));
      float4* d4 = (float4*)sm.u.encS;
      for (int i = lane; i < (MAXL * HIDD) / 4; i += 32) d4[i] = e4[i];
      const float4* h4 = (const float4*)(hid + row * HIDD);
      float4* hd = (float4*)(sm.hidH + r * HIDD);
      for (int i = lane; i < HIDD / 4; i += 32) hd[i] = h4[i];
    }
    __syncthreads();
    // 81 enc-dot tasks + 9 hidden-dot tasks over the 32 lanes
    for (int t = lane; t < 90; t += 32) {
      if (t < 81) {
        const int l = t / 9, m = t % 9;
        const float4* e4 = (const float4*)(sm.u.encS + l * HIDD);
        const float4* w4 = (const float4*)(aW + m * (2 * HIDD));
        float s = 0.f;
        for (int f = 0; f < HIDD / 4; ++f) {
          float4 a = e4[f], b = w4[f];
          s += a.x * b.x + a.y * b.y + a.z * b.z + a.w * b.w;
        }
        sm.s1[t] = s;
      } else {
        const int m = t - 81;
        const float4* h4 = (const float4*)(sm.hidH + r * HIDD);
        const float4* w4 = (const float4*)(aW + m * (2 * HIDD) + HIDD);
        float s = 0.f;
        for (int f = 0; f < HIDD / 4; ++f) {
          float4 a = h4[f], b = w4[f];
          s += a.x * b.x + a.y * b.y + a.z * b.z + a.w * b.w;
        }
        sm.s2[m] = s + aB[m];
      }
    }
    __syncthreads();
    if (lane < MAXL) {                       // softmax over m for row l=lane
      float sv[MAXL]; float mx = -3.4e38f;
      #pragma unroll
      for (int m = 0; m < MAXL; ++m) { sv[m] = sm.s1[lane * 9 + m] + sm.s2[m]; mx = fmaxf(mx, sv[m]); }
      float sum = 0.f;
      #pragma unroll
      for (int m = 0; m < MAXL; ++m) { sv[m] = __expf(sv[m] - mx); sum += sv[m]; }
      const float inv = 1.f / sum;
      #pragma unroll
      for (int m = 0; m < MAXL; ++m) sm.w[lane * 9 + m] = sv[m] * inv;
    }
    __syncthreads();
    if (lane < MAXL) {                       // alpha_m = sum_l w[l][m]
      float a = 0.f;
      #pragma unroll
      for (int l = 0; l < MAXL; ++l) a += sm.w[l * 9 + lane];
      sm.alpha[lane] = a;
    }
    __syncthreads();
    {
      const int h0 = lane * 8;               // context: 8 dims per lane
      float c[8] = {0,0,0,0,0,0,0,0};
      for (int m = 0; m < MAXL; ++m) {
        const float av = sm.alpha[m];
        const float* ev = sm.u.encS + m * HIDD + h0;
        #pragma unroll
        for (int j = 0; j < 8; ++j) c[j] += av * ev[j];
      }
      #pragma unroll
      for (int j = 0; j < 8; ++j) sm.xA[r * (EMBD + HIDD) + EMBD + h0 + j] = (__bf16)c[j];
      const int word = dw[row];              // embedding gather: 16 f32 per lane
      const float* er = emb + (long)word * EMBD + lane * 16;
      #pragma unroll
      for (int j = 0; j < 16; ++j) sm.xA[r * (EMBD + HIDD) + lane * 16 + j] = (__bf16)er[j];
    }
    __syncthreads();
  }

  // ---- Phase 2: x = relu([emb|ctx] @ combW^T + cb) -> x2[16][512] bf16
  for (int nt = 0; nt < EMBD / 16; ++nt) {
    vf8 acc = vzero8();
    const int col = nt * 16 + l15;
    const __bf16* wr = wc + (long)col * (EMBD + HIDD) + half * 8;
    const __bf16* ar = sm.xA + l15 * (EMBD + HIDD) + half * 8;
    for (int kt = 0; kt < (EMBD + HIDD) / 32; ++kt) {
      vbf16 a = load_a_bf16(ar + kt * 32);
      vbf16 b = load_b_bf16(wr + kt * 32);
      acc = WMMA_BF16(a, b, acc);
    }
    const float bias = cb[col];
    #pragma unroll
    for (int v = 0; v < 8; ++v) {
      float x = acc[v] + bias;
      x = x > 0.f ? x : 0.f;
      sm.u.x2[(half * 8 + v) * EMBD + col] = (__bf16)x;
    }
  }
  __syncthreads();

  // ---- Phase 3: GRU gates (r,z,n) per 16-col tile of HID
  for (int j = 0; j < HIDD / 16; ++j) {
    const int cr = j * 16 + l15;
    const int cz = cr + HIDD;
    const int cn = cr + 2 * HIDD;
    vf8 air = vzero8(), aiz = vzero8(), ain = vzero8();
    vf8 ahr = vzero8(), ahz = vzero8(), ahn = vzero8();
    {
      const __bf16* ar   = sm.u.x2 + l15 * EMBD + half * 8;
      const __bf16* wr_r = wih + (long)cr * EMBD + half * 8;
      const __bf16* wr_z = wih + (long)cz * EMBD + half * 8;
      const __bf16* wr_n = wih + (long)cn * EMBD + half * 8;
      for (int kt = 0; kt < EMBD / 32; ++kt) {
        vbf16 a = load_a_bf16(ar + kt * 32);
        air = WMMA_BF16(a, load_b_bf16(wr_r + kt * 32), air);
        aiz = WMMA_BF16(a, load_b_bf16(wr_z + kt * 32), aiz);
        ain = WMMA_BF16(a, load_b_bf16(wr_n + kt * 32), ain);
      }
    }
    {
      const float*  ar   = sm.hidH + l15 * HIDD + half * 8;
      const __bf16* wr_r = whh + (long)cr * HIDD + half * 8;
      const __bf16* wr_z = whh + (long)cz * HIDD + half * 8;
      const __bf16* wr_n = whh + (long)cn * HIDD + half * 8;
      for (int kt = 0; kt < HIDD / 32; ++kt) {
        vbf16 a = load_a_f32(ar + kt * 32);
        ahr = WMMA_BF16(a, load_b_bf16(wr_r + kt * 32), ahr);
        ahz = WMMA_BF16(a, load_b_bf16(wr_z + kt * 32), ahz);
        ahn = WMMA_BF16(a, load_b_bf16(wr_n + kt * 32), ahn);
      }
    }
    const float bir = bih[cr], biz = bih[cz], bin_ = bih[cn];
    const float bhr = bhh[cr], bhz = bhh[cz], bhn  = bhh[cn];
    #pragma unroll
    for (int v = 0; v < 8; ++v) {
      const int rv = half * 8 + v;
      const float rg = sigmoidf_(air[v] + bir + ahr[v] + bhr);
      const float zg = sigmoidf_(aiz[v] + biz + ahz[v] + bhz);
      const float ng = tanhf(ain[v] + bin_ + rg * (ahn[v] + bhn));
      const float hprev = sm.hidH[rv * HIDD + cr];
      const float hn = (1.f - zg) * ng + zg * hprev;
      const long grow = r0 + rv;
      hout[grow * HIDD + cr] = hn;
      hbf[grow * HIDD + cr] = (__bf16)hn;
    }
  }
}

// ---------------------------------------------------------------- vocab projection + log_softmax
__global__ __launch_bounds__(32) void out_logsoftmax(
    const __bf16* __restrict__ hbf, const __bf16* __restrict__ wout,
    const float* __restrict__ ob, float* __restrict__ out)
{
  const int lane = threadIdx.x;
  const int half = lane >> 4;
  const int l15  = lane & 15;
  const long r0  = (long)blockIdx.x * 16;

  vbf16 aF[8];                              // full K=256 A panel, register resident
  #pragma unroll
  for (int kt = 0; kt < 8; ++kt)
    aF[kt] = load_a_bf16(hbf + (r0 + l15) * HIDD + kt * 32 + half * 8);

  float lmax[8], lsum[8];                   // online logsumexp, per-lane column stream
  #pragma unroll
  for (int v = 0; v < 8; ++v) { lmax[v] = -__builtin_inff(); lsum[v] = 0.f; }

  const int NT = (VOCAB + 15) / 16;         // 256 N-tiles, last one partial (14 cols)
  for (int nt = 0; nt < NT; ++nt) {
    const int col = nt * 16 + l15;
    const int cc = col < VOCAB ? col : (VOCAB - 1);
    const bool valid = col < VOCAB;
    vf8 acc = vzero8();
    const __bf16* wr = wout + (long)cc * HIDD + half * 8;
    #pragma unroll
    for (int kt = 0; kt < 8; ++kt)
      acc = WMMA_BF16(aF[kt], load_b_bf16(wr + kt * 32), acc);
    const float bias = ob[cc];
    #pragma unroll
    for (int v = 0; v < 8; ++v) {
      const float t = valid ? (acc[v] + bias) : -__builtin_inff();
      const float m2 = fmaxf(lmax[v], t);
      lsum[v] = lsum[v] * __expf(lmax[v] - m2) + __expf(t - m2);
      lmax[v] = m2;
    }
  }
  // merge (max,sum) across the 16 lanes of each half-wave
  #pragma unroll
  for (int mask = 1; mask < 16; mask <<= 1) {
    #pragma unroll
    for (int v = 0; v < 8; ++v) {
      const float om = __shfl_xor(lmax[v], mask, 32);
      const float os = __shfl_xor(lsum[v], mask, 32);
      const float m2 = fmaxf(lmax[v], om);
      lsum[v] = lsum[v] * __expf(lmax[v] - m2) + os * __expf(om - m2);
      lmax[v] = m2;
    }
  }
  float lz[8];
  #pragma unroll
  for (int v = 0; v < 8; ++v) lz[v] = lmax[v] + __logf(lsum[v]);

  // pass 2: recompute logits from register-resident A and write log-probs
  for (int nt = 0; nt < NT; ++nt) {
    const int col = nt * 16 + l15;
    const int cc = col < VOCAB ? col : (VOCAB - 1);
    const bool valid = col < VOCAB;
    vf8 acc = vzero8();
    const __bf16* wr = wout + (long)cc * HIDD + half * 8;
    #pragma unroll
    for (int kt = 0; kt < 8; ++kt)
      acc = WMMA_BF16(aF[kt], load_b_bf16(wr + kt * 32), acc);
    const float bias = ob[cc];
    if (valid) {
      #pragma unroll
      for (int v = 0; v < 8; ++v)
        out[(r0 + half * 8 + v) * (long)VOCAB + col] = acc[v] + bias - lz[v];
    }
  }
}

// ---------------------------------------------------------------- host
extern "C" void kernel_launch(void* const* d_in, const int* in_sizes, int n_in,
                              void* d_out, int out_size, void* d_ws, size_t ws_size,
                              hipStream_t stream) {
  (void)in_sizes; (void)n_in; (void)out_size; (void)ws_size;
  const int*   dw    = (const int*)  d_in[0];
  const float* hid   = (const float*)d_in[1];
  const float* enc   = (const float*)d_in[2];
  const float* emb   = (const float*)d_in[3];
  const float* attnW = (const float*)d_in[4];
  const float* attnB = (const float*)d_in[5];
  const float* combW = (const float*)d_in[6];
  const float* combB = (const float*)d_in[7];
  const float* Wih   = (const float*)d_in[8];
  const float* bih   = (const float*)d_in[9];
  const float* Whh   = (const float*)d_in[10];
  const float* bhh   = (const float*)d_in[11];
  const float* outW  = (const float*)d_in[12];
  const float* outB  = (const float*)d_in[13];
  float* out = (float*)d_out;

  // workspace layout (bytes, all 16B aligned)
  char* ws = (char*)d_ws;
  __bf16* wc   = (__bf16*)(ws);                          // 512*768   *2 = 786432
  __bf16* wih  = (__bf16*)(ws + 786432);                 // 768*512   *2 = 786432
  __bf16* whh  = (__bf16*)(ws + 1572864);                // 768*256   *2 = 393216
  __bf16* wout = (__bf16*)(ws + 1966080);                // 4094*256  *2 = 2096128
  __bf16* hbf  = (__bf16*)(ws + 4062208);                // 32768*256 *2 = 16777216

  const int T = 256;
  cvt_bf16<<<(EMBD * (EMBD + HIDD) + T - 1) / T, T, 0, stream>>>(combW, wc,  EMBD * (EMBD + HIDD));
  cvt_bf16<<<(3 * HIDD * EMBD + T - 1) / T,     T, 0, stream>>>(Wih,   wih, 3 * HIDD * EMBD);
  cvt_bf16<<<(3 * HIDD * HIDD + T - 1) / T,     T, 0, stream>>>(Whh,   whh, 3 * HIDD * HIDD);
  cvt_bf16<<<(VOCAB * HIDD + T - 1) / T,        T, 0, stream>>>(outW,  wout, VOCAB * HIDD);

  float* hout = out + (size_t)BATCH * VOCAB;   // h_new follows the log-probs in d_out

  fused_attn_gru<<<BATCH / 16, 32, 0, stream>>>(dw, hid, enc, emb, attnW, attnB,
                                                wc, combB, wih, bih, whh, bhh,
                                                hout, hbf);
  out_logsoftmax<<<BATCH / 16, 32, 0, stream>>>(hbf, wout, outB, out);
}
